// RNN_59287728554007
// MI455X (gfx1250) — matrix operand order repaired and checked by reading, not compiled, not run on hardware
//
#include <hip/hip_runtime.h>

// ---- CDNA5 WMMA types -------------------------------------------------------
typedef __attribute__((ext_vector_type(16))) __bf16 v16bf;
typedef __attribute__((ext_vector_type(8)))  __bf16 v8bf;
typedef __attribute__((ext_vector_type(8)))  float  v8f;
// ---- TDM descriptor group types (clang-23 / therock 6-arg form) -------------
typedef __attribute__((ext_vector_type(4))) unsigned int v4u;
typedef __attribute__((ext_vector_type(8))) int          v8i_t;
typedef __attribute__((ext_vector_type(4))) int          v4i_t;

static __device__ __forceinline__ v8f wmma_bf16(v16bf a, v16bf b, v8f c) {
  // D = A(16x32 bf16) x B(32x16 bf16) + C(16x16 f32)
  return __builtin_amdgcn_wmma_f32_16x16x32_bf16(false, a, false, b,
                                                 (short)0, c, false, false);
}

constexpr int B_ = 64, T_ = 512, I_ = 256, H_ = 1024, O_ = 128;
constexpr int HP = 1040;   // padded row stride (bf16 elems): LDS bank spread,
                           // and identical layout in global so TDM copies are 1-D

#if __has_builtin(__builtin_amdgcn_tensor_load_to_lds)
#define HAVE_TDM 1
// 1-D contiguous DMA: global -> LDS, n8 elements of 8 bytes.
static __device__ __forceinline__ void tdm_load_1d(unsigned lds_off,
                                                   const void* gaddr,
                                                   unsigned n8) {
  unsigned long long ga = (unsigned long long)gaddr;
  v4u g0;
  g0[0] = 1u;                                   // count=1 (valid), load, no gather
  g0[1] = lds_off;                              // D#.lds_addr (bytes)
  g0[2] = (unsigned)(ga & 0xffffffffu);         // global_addr[31:0]
  g0[3] = (unsigned)((ga >> 32) & 0x1ffffffu)   // global_addr[56:32]
        | (2u << 30);                           // type = 2 ("image")
  v8i_t g1 = (v8i_t)0;
  g1[0] = (3 << 16);                            // workgroup_mask=0, data_size=3 (8B)
  g1[1] = (int)((n8 & 0xffffu) << 16);          // tensor_dim0[15:0]  (bits 63:48)
  g1[2] = (int)((n8 >> 16) & 0xffffu);          // tensor_dim0[31:16] (bits 79:64)
  g1[3] = (int)((n8 & 0xffffu) << 16);          // tile_dim0          (bits 127:112)
  v4i_t gz4 = (v4i_t)0;                         // groups 2/3 unused (1-D tensor)
  v8i_t gz8 = (v8i_t)0;                         // trailing group (6-arg form)
  __builtin_amdgcn_tensor_load_to_lds(g0, g1, gz4, gz4, gz8, 0);
}
#else
#define HAVE_TDM 0
#endif

// ---- small utility kernels --------------------------------------------------
__global__ void zero_u32_kernel(unsigned int* __restrict__ p, int n) {
  int i = blockIdx.x * blockDim.x + threadIdx.x;
  if (i < n) p[i] = 0u;
}

__global__ void f32_to_bf16_kernel(const float* __restrict__ src,
                                   __bf16* __restrict__ dst, int n) {
  int i = blockIdx.x * blockDim.x + threadIdx.x;
  if (i < n) dst[i] = (__bf16)src[i];
}

// W_hh f32 [H][H] -> padded bf16 image [H][HP] (pad cols never read).
__global__ void whh_pad_kernel(const float* __restrict__ src,
                               __bf16* __restrict__ dst) {
  int i = blockIdx.x * blockDim.x + threadIdx.x;
  if (i < H_ * H_) {
    int r = i >> 10, k = i & (H_ - 1);
    dst[(size_t)r * HP + k] = (__bf16)src[i];
  }
}

// ---- Phase 1: xp[t][b][h] = x[b,t,:] . W_ih[h,:] + b_ih[h] + b_hh[h] --------
// A operand comes from the pre-converted bf16 copy of x: inner loop is pure
// b128 loads + WMMA, no f32->bf16 converts on the critical path.
__global__ __launch_bounds__(128)
void xp_gemm_kernel(const __bf16* __restrict__ xb, const __bf16* __restrict__ Wih,
                    const float* __restrict__ b_ih, const float* __restrict__ b_hh,
                    float* __restrict__ xp) {
  const int bid   = blockIdx.x;
  const int hTile = bid & 15;          // 16 column-tiles of 64
  const int btile = (bid >> 4) & 3;    // 4 batch tiles of 16
  const int t     = bid >> 6;          // 0..511
  const int lane  = threadIdx.x & 31;
  const int wave  = threadIdx.x >> 5;  // 0..3
  const int hi    = lane >> 4;
  const int l15   = lane & 15;

  const int bglob = btile * 16 + l15;
  const __bf16* xrow = xb + ((size_t)bglob * T_ + t) * I_;
  const int ncol  = hTile * 64 + wave * 16 + l15;
  const __bf16* wrow = Wih + (size_t)ncol * I_;

  v8f acc = {};
  #pragma unroll
  for (int k = 0; k < I_; k += 32) {
    const int kA = k + hi * 16;
    v8bf alo = *(const v8bf*)(xrow + kA);
    v8bf ahi = *(const v8bf*)(xrow + kA + 8);
    v16bf a;
    #pragma unroll
    for (int i = 0; i < 8; i++) { a[i] = alo[i]; a[i + 8] = ahi[i]; }
    v16bf b = *(const v16bf*)(wrow + k + hi * 16);
    acc = wmma_bf16(a, b, acc);
  }

  const float bias = b_ih[ncol] + b_hh[ncol];
  #pragma unroll
  for (int r = 0; r < 8; r++) {
    const int m = hi * 8 + r;
    xp[((size_t)t * B_ + btile * 16 + m) * H_ + ncol] = acc[r] + bias;
  }
}

// ---- Phase 2: persistent scan ----------------------------------------------
// grid = 64 blocks = (4 batch tiles) x (16 H-slices of 64 cols), 256 thr (8 wv).
// W_hh slice (64xHP bf16 = 130 KB) is TDM-DMA'd into LDS once; each step the
// 16xHP h tile (32.5 KB) is TDM-DMA'd from the L2-resident ping-pong buffer.
__global__ __launch_bounds__(256)
void rnn_scan_kernel(const __bf16* __restrict__ WhhP,  // bf16 [H][HP]
                     const float* __restrict__ xp,     // f32 [T][B][H]
                     __bf16* __restrict__ hbuf0,       // bf16 [B][HP] ping
                     __bf16* __restrict__ hbuf1,       // bf16 [B][HP] pong
                     unsigned int* __restrict__ cnt)   // [T]
{
  __shared__ __align__(32) __bf16 w_lds[64][HP];
  __shared__ __align__(32) __bf16 h_lds[16][HP];
  __shared__ float partial[4][256];

  const int btile  = blockIdx.x & 3;
  const int hslice = blockIdx.x >> 2;   // 0..15
  const int tid  = threadIdx.x;
  const int lane = tid & 31;
  const int wave = tid >> 5;            // 0..7
  const int hi   = lane >> 4;
  const int l15  = lane & 15;

  const int tile  = wave & 3;           // n-tile within the 64-col slice
  const int khalf = wave >> 2;          // K split (0: k<512, 1: k>=512)
  const int colg  = hslice * 64 + tile * 16 + l15;

#if HAVE_TDM
  const unsigned w_lds_off = (unsigned)(unsigned long long)(void*)&w_lds[0][0];
  const unsigned h_lds_off = (unsigned)(unsigned long long)(void*)&h_lds[0][0];
  // One-time DMA of this block's W_hh slice: contiguous 64*HP bf16.
  if (wave == 0) {
    tdm_load_1d(w_lds_off, WhhP + (size_t)hslice * 64 * HP, 64 * HP / 4);
    __builtin_amdgcn_s_wait_tensorcnt(0);
  }
#else
  for (int e = tid; e < 64 * (HP / 8); e += 256) {
    const int c = e / (HP / 8), k8 = (e % (HP / 8)) * 8;
    *(v8bf*)&w_lds[c][k8] = *(const v8bf*)(WhhP + (size_t)(hslice * 64 + c) * HP + k8);
  }
#endif
  __syncthreads();

  __bf16* hb[2] = { hbuf0, hbuf1 };

  for (int t = 0; t < T_; t++) {
    const __bf16* hc = hb[t & 1];
    __bf16*       hn = hb[(t + 1) & 1];

#if HAVE_TDM
    // Async DMA the 16-row h tile (contiguous 16*HP bf16) into LDS.
    if (wave == 0) {
      tdm_load_1d(h_lds_off, hc + (size_t)btile * 16 * HP, 16 * HP / 4);
    }
#else
    for (int e = tid; e < 16 * (HP / 8); e += 256) {
      const int m = e / (HP / 8), k8 = (e % (HP / 8)) * 8;
      *(v8bf*)&h_lds[m][k8] = *(const v8bf*)(hc + (size_t)(btile * 16 + m) * HP + k8);
    }
#endif
    // Prefetch the xp values this block consumes at the end of the step.
    if (wave < 4) {
      #pragma unroll
      for (int r = 0; r < 8; r += 2)
        __builtin_prefetch(
            &xp[((size_t)t * B_ + btile * 16 + hi * 8 + r) * H_ + colg], 0, 0);
    }
#if HAVE_TDM
    if (wave == 0) __builtin_amdgcn_s_wait_tensorcnt(0);
#endif
    __syncthreads();

    // One 16x16 output tile over half of K (16 chained WMMAs).
    v8f acc = {};
    #pragma unroll 4
    for (int kt = 0; kt < 16; kt++) {
      const int k  = khalf * 512 + kt * 32;
      const int kA = k + hi * 16;
      v8bf alo = *(const v8bf*)&h_lds[l15][kA];
      v8bf ahi = *(const v8bf*)&h_lds[l15][kA + 8];
      v16bf a;
      #pragma unroll
      for (int i = 0; i < 8; i++) { a[i] = alo[i]; a[i + 8] = ahi[i]; }
      v16bf b = *(const v16bf*)&w_lds[tile * 16 + l15][k + hi * 16];
      acc = wmma_bf16(a, b, acc);
    }

    // Cross-wave K reduction: waves 4..7 dump partials, waves 0..3 combine.
    if (wave >= 4) {
      #pragma unroll
      for (int r = 0; r < 8; r++)
        partial[tile][(hi * 8 + r) * 16 + l15] = acc[r];
    }
    __syncthreads();
    if (wave < 4) {
      #pragma unroll
      for (int r = 0; r < 8; r++) {
        const int m = hi * 8 + r;
        const float v = acc[r] + partial[tile][m * 16 + l15]
                      + xp[((size_t)t * B_ + btile * 16 + m) * H_ + colg];
        hn[(size_t)(btile * 16 + m) * HP + colg] = (__bf16)tanhf(v);
      }
    }
    __threadfence();
    __syncthreads();   // also guards partial[]/h_lds reuse

    // Grid barrier: all 64 blocks must publish h(t+1) before anyone reads it.
    if (tid == 0) {
      __hip_atomic_fetch_add(&cnt[t], 1u, __ATOMIC_ACQ_REL,
                             __HIP_MEMORY_SCOPE_AGENT);
      while (__hip_atomic_load(&cnt[t], __ATOMIC_ACQUIRE,
                               __HIP_MEMORY_SCOPE_AGENT) < 64u)
        __builtin_amdgcn_s_sleep(2);
    }
    __syncthreads();
  }
  // final h (t=512) ends up in hbuf0
}

// ---- Phase 3: out = h_last @ W_lin^T + b_lin  (64x128, K=1024) --------------
__global__ __launch_bounds__(1024)
void final_linear_kernel(const __bf16* __restrict__ hlast,  // [B][HP]
                         const __bf16* __restrict__ Wlin,   // [O][H] bf16
                         const float* __restrict__ b_lin,
                         float* __restrict__ out) {
  const int tid  = threadIdx.x;
  const int lane = tid & 31;
  const int wave = tid >> 5;
  const int mb = wave >> 3, nb = wave & 7;
  const int hi = lane >> 4, l15 = lane & 15;

  const __bf16* arow = hlast + (size_t)(mb * 16 + l15) * HP;
  const int ocol = nb * 16 + l15;
  const __bf16* wrow = Wlin + (size_t)ocol * H_;

  v8f acc = {};
  #pragma unroll 4
  for (int k = 0; k < H_; k += 32) {
    const int kA = k + hi * 16;
    v8bf alo = *(const v8bf*)(arow + kA);
    v8bf ahi = *(const v8bf*)(arow + kA + 8);
    v16bf a;
    #pragma unroll
    for (int i = 0; i < 8; i++) { a[i] = alo[i]; a[i + 8] = ahi[i]; }
    v16bf b = *(const v16bf*)(wrow + k + hi * 16);
    acc = wmma_bf16(a, b, acc);
  }

  const float bias = b_lin[ocol];
  #pragma unroll
  for (int r = 0; r < 8; r++) {
    const int m = hi * 8 + r;
    out[(size_t)(mb * 16 + m) * O_ + ocol] = acc[r] + bias;
  }
}

// ---- host-side orchestration ------------------------------------------------
extern "C" void kernel_launch(void* const* d_in, const int* in_sizes, int n_in,
                              void* d_out, int out_size, void* d_ws, size_t ws_size,
                              hipStream_t stream) {
  const float* x     = (const float*)d_in[0];
  const float* W_ih  = (const float*)d_in[1];
  const float* W_hh  = (const float*)d_in[2];
  const float* b_ih  = (const float*)d_in[3];
  const float* b_hh  = (const float*)d_in[4];
  const float* W_lin = (const float*)d_in[5];
  const float* b_lin = (const float*)d_in[6];
  float* out = (float*)d_out;

  // Workspace layout (all regions 32B aligned):
  //   Wih_b   : H*I    bf16  (512 KB)
  //   Wlin_b  : O*H    bf16  (256 KB)
  //   Whh_pad : H*HP   bf16  (2.1 MB, padded image for 1-D TDM slices)
  //   hbuf0/1 : B*HP   bf16  (130 KB each, padded ping-pong hidden state)
  //   xb      : B*T*I  bf16  (16.8 MB, bf16 copy of x)
  //   cnt     : T      u32   (grid-barrier counters)
  //   xp      : T*B*H  f32   (128 MB input projections)
  __bf16* Wih_b   = (__bf16*)d_ws;
  __bf16* Wlin_b  = Wih_b + (size_t)H_ * I_;
  __bf16* Whh_pad = Wlin_b + (size_t)O_ * H_;
  __bf16* hbuf0   = Whh_pad + (size_t)H_ * HP;
  __bf16* hbuf1   = hbuf0 + (size_t)B_ * HP;
  __bf16* xb      = hbuf1 + (size_t)B_ * HP;
  unsigned int* cnt = (unsigned int*)(xb + (size_t)B_ * T_ * I_);
  float* xp = (float*)(cnt + T_);

  // h0 = 0 (incl. pad cols) and barrier counters = 0.
  zero_u32_kernel<<<(B_ * HP / 2 + 255) / 256, 256, 0, stream>>>(
      (unsigned int*)hbuf0, B_ * HP / 2);
  zero_u32_kernel<<<(T_ + 255) / 256, 256, 0, stream>>>(cnt, T_);

  // Precision conversions (bf16 operands for WMMA).
  f32_to_bf16_kernel<<<(H_ * I_ + 255) / 256, 256, 0, stream>>>(W_ih, Wih_b, H_ * I_);
  f32_to_bf16_kernel<<<(O_ * H_ + 255) / 256, 256, 0, stream>>>(W_lin, Wlin_b, O_ * H_);
  f32_to_bf16_kernel<<<(B_ * T_ * I_ + 255) / 256, 256, 0, stream>>>(
      x, xb, B_ * T_ * I_);
  whh_pad_kernel<<<(H_ * H_ + 255) / 256, 256, 0, stream>>>(W_hh, Whh_pad);

  // Phase 1: big WMMA GEMM for all timesteps' input projections.
  xp_gemm_kernel<<<T_ * 4 * 16, 128, 0, stream>>>(xb, Wih_b, b_ih, b_hh, xp);

  // Phase 2: persistent recurrent scan, W_hh resident in LDS via TDM.
  rnn_scan_kernel<<<64, 256, 0, stream>>>(Whh_pad, xp, hbuf0, hbuf1, cnt);

  // Phase 3: final projection.
  final_linear_kernel<<<1, 1024, 0, stream>>>(hbuf0, Wlin_b, b_lin, out);
}
